// MixTemporalGNN_32899449487585
// MI455X (gfx1250) — compile-verified, matrix operand order checked by reading, not run on hardware
//
#include <hip/hip_runtime.h>
#include <cstdint>
#include <cstddef>

typedef __attribute__((ext_vector_type(16))) __bf16 v16bf;
typedef __attribute__((ext_vector_type(8)))  __bf16 v8bf;
typedef __attribute__((ext_vector_type(8)))  float  v8f;
typedef __attribute__((ext_vector_type(4)))  float  v4f;

#define DEV __device__ __forceinline__

static inline unsigned cdivu(long long a, long long b) { return (unsigned)((a + b - 1) / b); }

#if __has_builtin(__builtin_amdgcn_tensor_load_to_lds) && __has_builtin(__builtin_amdgcn_s_wait_tensorcnt)
#define HAVE_TDM 1
#else
#define HAVE_TDM 0
#endif

// ---------------------------------------------------------------------------
// CDNA5 WMMA bf16 fragment helpers (wave32, 16x16x32, f32 accumulate)
// A fragment (16x32, bf16): lane L holds row m = L&15; element i holds
//   K = kb + (i/8)*16 + (L>>4)*8 + (i%8)  -> two contiguous 8-elt runs.
// B fragment (32x16): same K pattern, lane L holds column n = L&15, sourced
//   from Bt stored as [N,K] row-major (i.e. B transposed).
// C/D fragment: element j of lane L is row (L>>4)*8 + j, col L&15.
// ---------------------------------------------------------------------------

DEV v16bf load_frag_bf(const __bf16* __restrict__ base, int ld, int row0, int kb, int lane) {
    int m = lane & 15, g = (lane >> 4) & 1;
    const __bf16* p = base + (size_t)(row0 + m) * (size_t)ld + (size_t)(kb + g * 8);
    v8bf lo = *(const v8bf*)p;
    v8bf hi = *(const v8bf*)(p + 16);
    v16bf r;
    for (int i = 0; i < 8; ++i) { r[i] = lo[i]; r[i + 8] = hi[i]; }
    return r;
}

DEV v16bf load_frag_f32_scaled(const float* __restrict__ base, int ld, int row0, int kb,
                               int lane, float scale) {
    int m = lane & 15, g = (lane >> 4) & 1;
    const float* p = base + (size_t)(row0 + m) * (size_t)ld + (size_t)(kb + g * 8);
    v4f a0 = *(const v4f*)p;
    v4f a1 = *(const v4f*)(p + 4);
    v4f b0 = *(const v4f*)(p + 16);
    v4f b1 = *(const v4f*)(p + 20);
    v16bf r;
    for (int i = 0; i < 4; ++i) {
        r[i]      = (__bf16)(a0[i] * scale);
        r[i + 4]  = (__bf16)(a1[i] * scale);
        r[i + 8]  = (__bf16)(b0[i] * scale);
        r[i + 12] = (__bf16)(b1[i] * scale);
    }
    return r;
}

DEV v8f wmma_bf16(v16bf a, v16bf b, v8f acc) {
    return __builtin_amdgcn_wmma_f32_16x16x32_bf16(false, a, false, b, (short)0, acc,
                                                   false, false);
}

DEV float sigmoidf(float x) { return 1.0f / (1.0f + expf(-x)); }

// ---------------------------------------------------------------------------
// Tensor Data Mover: 2D tile (rows x cols) global -> LDS.  D# per ISA ch.8.
// dsize_code: 0=1B 1=2B 2=4B.  stride0/tensor dims in elements.
// ---------------------------------------------------------------------------
#if HAVE_TDM
typedef __attribute__((ext_vector_type(4))) unsigned int tdm_u32x4;
typedef __attribute__((ext_vector_type(8))) int          tdm_i32x8;
typedef __attribute__((ext_vector_type(4))) int          tdm_i32x4;

DEV void tdm_load_2d(unsigned lds_off, const void* gptr, unsigned dsize_code,
                     unsigned tile_d0, unsigned tile_d1,
                     unsigned long long stride0, unsigned tensor_d0, unsigned tensor_d1) {
    unsigned long long ga = (unsigned long long)(size_t)gptr;
    tdm_u32x4 g0;
    g0[0] = 1u;                                        // count=1, user descriptor
    g0[1] = lds_off;                                   // lds_addr (bytes)
    g0[2] = (unsigned)ga;                              // global_addr[31:0]
    g0[3] = (unsigned)((ga >> 32) & 0x1FFFFFFu) | (2u << 30);  // addr[56:32] | type=2
    tdm_i32x8 g1;
    g1[0] = (int)(dsize_code << 16);                   // data_size
    g1[1] = (int)((tensor_d0 & 0xFFFFu) << 16);        // tensor_dim0[15:0] @ [63:48]
    g1[2] = (int)((tensor_d0 >> 16) | ((tensor_d1 & 0xFFFFu) << 16));
    g1[3] = (int)((tensor_d1 >> 16) | ((tile_d0 & 0xFFFFu) << 16));
    g1[4] = (int)(tile_d1 & 0xFFFFu);                  // tile_dim1 (tile_dim2=0)
    g1[5] = (int)(stride0 & 0xFFFFFFFFull);            // tensor_dim0_stride[31:0]
    g1[6] = (int)((stride0 >> 32) & 0xFFFFull);        // stride[47:32] (dim1_stride=0)
    g1[7] = 0;
    tdm_i32x4 z4 = {};
#if __clang_major__ >= 23
    tdm_i32x8 z8 = {};
    __builtin_amdgcn_tensor_load_to_lds(g0, g1, z4, z4, z8, 0);
#else
    __builtin_amdgcn_tensor_load_to_lds(g0, g1, z4, z4, 0);
#endif
}
#endif  // HAVE_TDM

// ---------------------------------------------------------------------------
// Generic WMMA GEMM: out = A[M,K](bf16) @ Bt[N,K](bf16)^T  (+ Cinit + bias)
// blockDim = 128 (4 waves); block -> 16 rows x 256 cols; wave -> 16x64 strip.
// A slab (16 x K) is TDM-staged to LDS once per block.
// Bt must have ceil(N/64)*64 valid (zero-padded) rows.
// ---------------------------------------------------------------------------

#define EP_BIAS_F32       0   // outF = acc + bias + (Cinit?)
#define EP_PRELU_ROWP_BF  1   // PReLU slope alpha[row % 15] -> outB
#define EP_SIGMUL_PACK_BF 2   // sigmoid(acc+bias)*other[r,c] -> outB[(r%15)*64+r/15]
#define EP_PRELU_COL_BF   3   // PReLU slope alpha[col] -> outB

__global__ void __launch_bounds__(128)
k_gemm(const __bf16* __restrict__ A, int lda,
       const __bf16* __restrict__ Bt, int ldb,
       int M, int N, int K,
       const float* __restrict__ bias,
       const float* __restrict__ Cinit, int ldci,
       float* __restrict__ outF, int ldo,
       __bf16* __restrict__ outB, int ldob,
       const float* __restrict__ alpha,
       const float* __restrict__ other, int ldoth,
       int mode) {
    extern __shared__ __bf16 sA[];   // [16][K]
    int tid = threadIdx.x, wave = tid >> 5, lane = tid & 31;
    int rt = blockIdx.x * 16;

#if HAVE_TDM
    if (wave == 0) {
        tdm_load_2d((unsigned)(size_t)sA, A + (size_t)rt * (size_t)lda, 1,
                    (unsigned)K, 16u, (unsigned long long)lda, (unsigned)lda, (unsigned)M);
        __builtin_amdgcn_s_wait_tensorcnt(0);
    }
    __syncthreads();
#else
    for (int e = tid * 8; e < 16 * K; e += 128 * 8) {
        int r = e / K, c = e % K;
        *(v8bf*)(sA + e) = *(const v8bf*)(A + (size_t)(rt + r) * (size_t)lda + c);
    }
    __syncthreads();
#endif

    int cn0 = (blockIdx.y * 4 + wave) * 64;
    if (cn0 >= N || rt >= M) return;

    v8f acc[4] = {};
    for (int kb = 0; kb < K; kb += 32) {
        v16bf a = load_frag_bf(sA, K, 0, kb, lane);
        const __bf16* pf = Bt + (size_t)(cn0 + (lane & 15)) * (size_t)ldb + kb +
                           ((lane >> 4) & 1) * 8 + 32;
        __builtin_prefetch((const void*)pf, 0, 1);
        v16bf b0 = load_frag_bf(Bt, ldb, cn0,      kb, lane);
        v16bf b1 = load_frag_bf(Bt, ldb, cn0 + 16, kb, lane);
        v16bf b2 = load_frag_bf(Bt, ldb, cn0 + 32, kb, lane);
        v16bf b3 = load_frag_bf(Bt, ldb, cn0 + 48, kb, lane);
        acc[0] = wmma_bf16(a, b0, acc[0]);
        acc[1] = wmma_bf16(a, b1, acc[1]);
        acc[2] = wmma_bf16(a, b2, acc[2]);
        acc[3] = wmma_bf16(a, b3, acc[3]);
    }

    int r0 = rt + ((lane >> 4) & 1) * 8;
    for (int t = 0; t < 4; ++t) {
        int c = cn0 + t * 16 + (lane & 15);
        if (c >= N) continue;
        for (int j = 0; j < 8; ++j) {
            int r = r0 + j;
            float z = acc[t][j];
            if (bias)  z += bias[c];
            if (Cinit) z += Cinit[(size_t)r * (size_t)ldci + c];
            if (mode == EP_BIAS_F32) {
                outF[(size_t)r * (size_t)ldo + c] = z;
            } else if (mode == EP_PRELU_ROWP_BF) {
                float al = alpha[r % 15];
                z = (z >= 0.0f) ? z : al * z;
                outB[(size_t)r * (size_t)ldob + c] = (__bf16)z;
            } else if (mode == EP_SIGMUL_PACK_BF) {
                float v = sigmoidf(z) * other[(size_t)r * (size_t)ldoth + c];
                int rr = (r % 15) * 64 + (r / 15);
                outB[(size_t)rr * (size_t)ldob + c] = (__bf16)v;
            } else { // EP_PRELU_COL_BF
                float al = alpha[c];
                z = (z >= 0.0f) ? z : al * z;
                outB[(size_t)r * (size_t)ldob + c] = (__bf16)z;
            }
        }
    }
}

// ---------------------------------------------------------------------------
// Fused SAGEConv layer: r = h@Ws + (neigh/deg)@Wn + b -> PReLU -> BatchNorm
// blockDim = 64 (2 waves); block -> 16 rows x 128 cols; wave -> 16x64 strip.
// A slab (bf16) and neigh slab (f32) TDM-staged to LDS.
// ---------------------------------------------------------------------------

__global__ void __launch_bounds__(64)
k_sage(const __bf16* __restrict__ A, int lda,
       const float* __restrict__ neigh, int Kd,
       const float* __restrict__ invdeg,
       const __bf16* __restrict__ Wst, const __bf16* __restrict__ Wnt,
       const float* __restrict__ bias, const float* __restrict__ alpha,
       const float* __restrict__ gamma, const float* __restrict__ beta,
       const float* __restrict__ mu, const float* __restrict__ var,
       __bf16* __restrict__ out, int Nn) {
    __shared__ __bf16 sA[16 * 128];
    __shared__ float  sN[16 * 128];
    int tid = threadIdx.x, wave = tid >> 5, lane = tid & 31;
    int rt = blockIdx.x * 16;

#if HAVE_TDM
    if (wave == 0) {
        tdm_load_2d((unsigned)(size_t)sA, A + (size_t)rt * (size_t)lda, 1,
                    (unsigned)Kd, 16u, (unsigned long long)lda, (unsigned)lda, (unsigned)Nn);
        tdm_load_2d((unsigned)(size_t)sN, neigh + (size_t)rt * (size_t)Kd, 2,
                    (unsigned)Kd, 16u, (unsigned long long)Kd, (unsigned)Kd, (unsigned)Nn);
        __builtin_amdgcn_s_wait_tensorcnt(0);
    }
    __syncthreads();
#else
    for (int e = tid * 8; e < 16 * Kd; e += 64 * 8) {
        int r = e / Kd, c = e % Kd;
        *(v8bf*)(sA + e) = *(const v8bf*)(A + (size_t)(rt + r) * (size_t)lda + c);
    }
    for (int e = tid * 4; e < 16 * Kd; e += 64 * 4) {
        int r = e / Kd, c = e % Kd;
        *(v4f*)(sN + e) = *(const v4f*)(neigh + (size_t)(rt + r) * (size_t)Kd + c);
    }
    __syncthreads();
#endif

    int cn0 = wave * 64;
    float sc = invdeg[rt + (lane & 15)];

    v8f acc[4] = {};
    for (int kb = 0; kb < Kd; kb += 32) {
        v16bf a1 = load_frag_bf(sA, Kd, 0, kb, lane);
        v16bf a2 = load_frag_f32_scaled(sN, Kd, 0, kb, lane, sc);
        for (int t = 0; t < 4; ++t) {
            v16bf b1 = load_frag_bf(Wst, Kd, cn0 + t * 16, kb, lane);
            v16bf b2 = load_frag_bf(Wnt, Kd, cn0 + t * 16, kb, lane);
            acc[t] = wmma_bf16(a1, b1, acc[t]);
            acc[t] = wmma_bf16(a2, b2, acc[t]);
        }
    }

    int r0 = rt + ((lane >> 4) & 1) * 8;
    for (int t = 0; t < 4; ++t) {
        int c = cn0 + t * 16 + (lane & 15);
        float bi = bias[c], al = alpha[c];
        float bn = rsqrtf(var[c] + 1e-5f) * gamma[c];
        float mb = mu[c], be = beta[c];
        for (int j = 0; j < 8; ++j) {
            int r = r0 + j;
            float z = acc[t][j] + bi;
            z = (z >= 0.0f) ? z : al * z;
            z = (z - mb) * bn + be;
            out[(size_t)r * 512 + c] = (__bf16)z;
        }
    }
}

// ---------------------------------------------------------------------------
// Small utility kernels
// ---------------------------------------------------------------------------

__global__ void k_zero_f32(float* __restrict__ p, long long n) {
    long long i = (long long)blockIdx.x * blockDim.x + threadIdx.x;
    if (i < n) p[i] = 0.0f;
}
__global__ void k_zero_bf16(__bf16* __restrict__ p, long long n) {
    long long i = (long long)blockIdx.x * blockDim.x + threadIdx.x;
    if (i < n) p[i] = (__bf16)0.0f;
}
__global__ void k_f32_to_bf16(const float* __restrict__ s, __bf16* __restrict__ d, long long n) {
    long long i = (long long)blockIdx.x * blockDim.x + threadIdx.x;
    if (i < n) d[i] = (__bf16)s[i];
}
// W[K,N] f32 -> Wt[N,K] bf16
__global__ void k_conv_transpose(const float* __restrict__ W, __bf16* __restrict__ Wt,
                                 int K, int N) {
    long long i = (long long)blockIdx.x * blockDim.x + threadIdx.x;
    if (i >= (long long)K * N) return;
    int k = (int)(i / N), n = (int)(i % N);
    Wt[(size_t)n * K + k] = (__bf16)W[i];
}
__global__ void k_embed(const int* __restrict__ tok, const float* __restrict__ emb,
                        __bf16* __restrict__ out, long long n64) {
    long long i = (long long)blockIdx.x * blockDim.x + threadIdx.x;
    if (i >= n64) return;
    int n = (int)(i >> 6), d = (int)(i & 63);
    out[i] = (__bf16)emb[(size_t)tok[n] * 64 + d];
}
__global__ void k_count(const int* __restrict__ idx, float* __restrict__ cnt, int n) {
    int i = blockIdx.x * blockDim.x + threadIdx.x;
    if (i < n) atomicAdd(&cnt[idx[i]], 1.0f);
}
__global__ void k_invclip(float* __restrict__ d, int n) {
    int i = blockIdx.x * blockDim.x + threadIdx.x;
    if (i < n) d[i] = 1.0f / fmaxf(d[i], 1.0f);
}
// neigh[dst[e]][d] += h[src[e]*ldh + d]
__global__ void k_edge_scatter(const __bf16* __restrict__ h, int ldh,
                               const int* __restrict__ src, const int* __restrict__ dst,
                               float* __restrict__ neigh, long long E, int D) {
    long long i = (long long)blockIdx.x * blockDim.x + threadIdx.x;
    if (i >= E * D) return;
    int e = (int)(i / D), d = (int)(i % D);
    atomicAdd(&neigh[(size_t)dst[e] * D + d], (float)h[(size_t)src[e] * ldh + d]);
}
__global__ void k_group_scatter(const __bf16* __restrict__ hc, const int* __restrict__ gid,
                                float* __restrict__ gv, long long n512) {
    long long i = (long long)blockIdx.x * blockDim.x + threadIdx.x;
    if (i >= n512) return;
    int n = (int)(i >> 9), c = (int)(i & 511);
    atomicAdd(&gv[(size_t)gid[n] * 512 + c], (float)hc[i]);
}
__global__ void k_group_fin(float* __restrict__ gv, const float* __restrict__ cnt,
                            __bf16* __restrict__ gvb) {
    int i = blockIdx.x * blockDim.x + threadIdx.x;
    if (i >= 960 * 512) return;
    float v = gv[i] / fmaxf(cnt[i >> 9], 1.0f);
    gv[i] = v;
    gvb[i] = (__bf16)v;
}
// LSTM pointwise cell update. gates[64,4096] in PyTorch i,f,g,o order.
__global__ void k_lstm_gate(const float* __restrict__ gates,
                            float* __restrict__ c, __bf16* __restrict__ hbf,
                            __bf16* __restrict__ ybf, int ldy) {
    int i = blockIdx.x * blockDim.x + threadIdx.x;
    if (i >= 64 * 1024) return;
    int b = i >> 10, j = i & 1023;
    const float* g4 = gates + (size_t)b * 4096;
    float gi = sigmoidf(g4[j]);
    float gf = sigmoidf(g4[1024 + j]);
    float gg = tanhf(g4[2048 + j]);
    float go = sigmoidf(g4[3072 + j]);
    float cc = gf * c[i] + gi * gg;
    float hh = go * tanhf(cc);
    c[i] = cc;
    hbf[i] = (__bf16)hh;
    if (ybf) ybf[(size_t)b * ldy + j] = (__bf16)hh;
}

// ---------------------------------------------------------------------------
// Host orchestration
// ---------------------------------------------------------------------------

extern "C" void kernel_launch(void* const* d_in, const int* in_sizes, int n_in,
                              void* d_out, int out_size, void* d_ws, size_t ws_size,
                              hipStream_t stream) {
    (void)in_sizes; (void)n_in; (void)out_size; (void)ws_size;

    const int Nh = 61440, Eh = 491520, Np = 245760, Ep = 983040;

    // ---- input pointers (setup_inputs dict order) ----
    auto I = [&](int i) { return d_in[i]; };
    const int  *h_tok = (const int*)I(0),  *h_src = (const int*)I(1);
    const int  *h_dst = (const int*)I(2),  *h_gid = (const int*)I(3);
    const float*h_emb = (const float*)I(4), *h_Ws1 = (const float*)I(5), *h_Wn1 = (const float*)I(6);
    const float*h_Wsr = (const float*)I(7), *h_Wnr = (const float*)I(8);
    const float*h_b   = (const float*)I(9), *h_a = (const float*)I(10);
    const float*h_bng = (const float*)I(11), *h_bnb = (const float*)I(12);
    const float*h_bnm = (const float*)I(13), *h_bnv = (const float*)I(14);
    const int  *p_tok = (const int*)I(15), *p_src = (const int*)I(16);
    const int  *p_dst = (const int*)I(17), *p_gid = (const int*)I(18);
    const float*p_emb = (const float*)I(19), *p_Ws1 = (const float*)I(20), *p_Wn1 = (const float*)I(21);
    const float*p_Wsr = (const float*)I(22), *p_Wnr = (const float*)I(23);
    const float*p_b   = (const float*)I(24), *p_a = (const float*)I(25);
    const float*p_bng = (const float*)I(26), *p_bnb = (const float*)I(27);
    const float*p_bnm = (const float*)I(28), *p_bnv = (const float*)I(29);
    const float*f1_W1 = (const float*)I(30), *f1_b1 = (const float*)I(31), *f1_a = (const float*)I(32);
    const float*f1_W2 = (const float*)I(33), *f1_b2 = (const float*)I(34);
    const float*f2_W1 = (const float*)I(35), *f2_b1 = (const float*)I(36), *f2_a = (const float*)I(37);
    const float*f2_W2 = (const float*)I(38), *f2_b2 = (const float*)I(39);
    const float*Wih0  = (const float*)I(40), *Whh0 = (const float*)I(41);
    const float*bih0  = (const float*)I(42), *bhh0 = (const float*)I(43);
    const float*Wih1  = (const float*)I(44), *Whh1 = (const float*)I(45);
    const float*bih1  = (const float*)I(46), *bhh1 = (const float*)I(47);
    const float*Wfc   = (const float*)I(48), *bfc = (const float*)I(49), *a_fc = (const float*)I(50);
    const float*Wc    = (const float*)I(51), *bc  = (const float*)I(52);

    // ---- workspace arena ----
    char* wsb = (char*)d_ws;
    size_t off = 0;
    auto alloc = [&](size_t bytes) -> void* {
        void* p = wsb + off;
        off = (off + bytes + 255) & ~(size_t)255;
        return p;
    };
    auto abf = [&](size_t elems) { return (__bf16*)alloc(elems * 2); };
    auto af  = [&](size_t elems) { return (float*)alloc(elems * 4); };

    __bf16 *hWs1t = abf(128 * 64),  *hWn1t = abf(128 * 64);
    __bf16 *hWsrt = abf(3 * 128 * 128), *hWnrt = abf(3 * 128 * 128);
    __bf16 *pWs1t = abf(128 * 64),  *pWn1t = abf(128 * 64);
    __bf16 *pWsrt = abf(3 * 128 * 128), *pWnrt = abf(3 * 128 * 128);
    __bf16 *f1W1t = abf(512 * 512), *f1W2t = abf(512 * 512);
    __bf16 *f2W1t = abf(512 * 512), *f2W2t = abf(512 * 512);
    __bf16 *Wih0b = abf((size_t)2 * 4096 * 1024), *Whh0b = abf((size_t)2 * 4096 * 1024);
    __bf16 *Wih1b = abf((size_t)2 * 4096 * 2048), *Whh1b = abf((size_t)2 * 4096 * 1024);
    __bf16 *Wfct  = abf(512 * 2048), *Wct = abf(128 * 512);   // Wct zero-padded to 128 rows
    float  *gv_h = af(960 * 512), *gv_p = af(960 * 512);
    __bf16 *gvh_bf = abf(960 * 512), *gvp_bf = abf(960 * 512);
    __bf16 *zf_bf  = abf(960 * 512);
    __bf16 *x_bf   = abf((size_t)960 * 1024);
    float  *xW     = af((size_t)960 * 4096);
    __bf16 *y1_bf  = abf((size_t)960 * 2048);
    float  *gates  = af(64 * 4096);
    float  *cst    = af(64 * 1024);
    __bf16 *h_bf   = abf(64 * 1024);
    __bf16 *hcat_bf = abf(64 * 2048);
    __bf16 *f_bf   = abf(64 * 512);
    float  *cnt    = af(960);
    float  *invdeg = af(Np);
    __bf16 *h0_bf  = abf((size_t)Np * 64);
    float  *neigh  = af((size_t)Np * 128);
    __bf16 *hc_bf  = abf((size_t)Np * 512);

    auto zf = [&](float* p, long long n) { k_zero_f32<<<cdivu(n, 256), 256, 0, stream>>>(p, n); };
    auto zb = [&](__bf16* p, long long n) { k_zero_bf16<<<cdivu(n, 256), 256, 0, stream>>>(p, n); };
    auto conv = [&](const float* s, __bf16* d, long long n) {
        k_f32_to_bf16<<<cdivu(n, 256), 256, 0, stream>>>(s, d, n);
    };
    auto convT = [&](const float* W, __bf16* Wt, int K, int N) {
        k_conv_transpose<<<cdivu((long long)K * N, 256), 256, 0, stream>>>(W, Wt, K, N);
    };
    auto gemm = [&](const __bf16* A, int lda, const __bf16* Bt, int ldb,
                    int M, int N, int K, const float* bias,
                    const float* Ci, int ldci, float* oF, int ldo,
                    __bf16* oB, int ldob, const float* al,
                    const float* oth, int ldoth, int mode) {
        dim3 g(cdivu(M, 16), cdivu(N, 256));
        size_t shmem = (size_t)16 * (size_t)K * 2;
        k_gemm<<<g, 128, shmem, stream>>>(A, lda, Bt, ldb, M, N, K, bias, Ci, ldci,
                                          oF, ldo, oB, ldob, al, oth, ldoth, mode);
    };

    // ---- phase 0: weight conversion ----
    convT(h_Ws1, hWs1t, 64, 128); convT(h_Wn1, hWn1t, 64, 128);
    convT(p_Ws1, pWs1t, 64, 128); convT(p_Wn1, pWn1t, 64, 128);
    for (int i = 0; i < 3; ++i) {
        convT(h_Wsr + i * 16384, hWsrt + i * 16384, 128, 128);
        convT(h_Wnr + i * 16384, hWnrt + i * 16384, 128, 128);
        convT(p_Wsr + i * 16384, pWsrt + i * 16384, 128, 128);
        convT(p_Wnr + i * 16384, pWnrt + i * 16384, 128, 128);
    }
    convT(f1_W1, f1W1t, 512, 512); convT(f1_W2, f1W2t, 512, 512);
    convT(f2_W1, f2W1t, 512, 512); convT(f2_W2, f2W2t, 512, 512);
    conv(Wih0, Wih0b, (long long)2 * 4096 * 1024);   // [out,in] == [N,K] already
    conv(Whh0, Whh0b, (long long)2 * 4096 * 1024);
    conv(Wih1, Wih1b, (long long)2 * 4096 * 2048);
    conv(Whh1, Whh1b, (long long)2 * 4096 * 1024);
    convT(Wfc, Wfct, 2048, 512);
    zb(Wct, 128 * 512);                    // zero-pad rows 120..127
    convT(Wc, Wct, 512, 120);

    // ---- phase 1: GCN encoders ----
    auto run_gcn = [&](int Nn, int E, const int* tok, const int* src, const int* dst,
                       const int* gid, const float* emb,
                       const __bf16* ws1t, const __bf16* wn1t,
                       const __bf16* wsrt, const __bf16* wnrt,
                       const float* bv, const float* av, const float* gvv,
                       const float* bev, const float* muv, const float* varv,
                       float* gv, __bf16* gvb) {
        zf(invdeg, Nn);
        k_count<<<cdivu(E, 256), 256, 0, stream>>>(dst, invdeg, E);
        k_invclip<<<cdivu(Nn, 256), 256, 0, stream>>>(invdeg, Nn);
        k_embed<<<cdivu((long long)Nn * 64, 256), 256, 0, stream>>>(tok, emb, h0_bf,
                                                                    (long long)Nn * 64);
        for (int i = 0; i < 4; ++i) {
            int Kd = (i == 0) ? 64 : 128;
            const __bf16* Ap = (i == 0) ? h0_bf : hc_bf + (size_t)(i - 1) * 128;
            int lda = (i == 0) ? 64 : 512;
            zf(neigh, (long long)Nn * Kd);
            k_edge_scatter<<<cdivu((long long)E * Kd, 256), 256, 0, stream>>>(
                Ap, lda, src, dst, neigh, E, Kd);
            const __bf16* Wst = (i == 0) ? ws1t : wsrt + (size_t)(i - 1) * 16384;
            const __bf16* Wnt = (i == 0) ? wn1t : wnrt + (size_t)(i - 1) * 16384;
            k_sage<<<cdivu(Nn, 16), 64, 0, stream>>>(Ap, lda, neigh, Kd, invdeg, Wst, Wnt,
                                          bv + i * 128, av + i * 128, gvv + i * 128,
                                          bev + i * 128, muv + i * 128, varv + i * 128,
                                          hc_bf + (size_t)i * 128, Nn);
        }
        zf(cnt, 960);
        k_count<<<cdivu(Nn, 256), 256, 0, stream>>>(gid, cnt, Nn);
        zf(gv, 960 * 512);
        k_group_scatter<<<cdivu((long long)Nn * 512, 256), 256, 0, stream>>>(
            hc_bf, gid, gv, (long long)Nn * 512);
        k_group_fin<<<cdivu(960 * 512, 256), 256, 0, stream>>>(gv, cnt, gvb);
    };
    run_gcn(Nh, Eh, h_tok, h_src, h_dst, h_gid, h_emb, hWs1t, hWn1t, hWsrt, hWnrt,
            h_b, h_a, h_bng, h_bnb, h_bnm, h_bnv, gv_h, gvh_bf);
    run_gcn(Np, Ep, p_tok, p_src, p_dst, p_gid, p_emb, pWs1t, pWn1t, pWsrt, pWnrt,
            p_b, p_a, p_bng, p_bnb, p_bnm, p_bnv, gv_p, gvp_bf);

    // ---- phase 2: cross-gated filters -> x[P,B,1024] ----
    gemm(gvh_bf, 512, f1W1t, 512, 960, 512, 512, f1_b1, nullptr, 0,
         nullptr, 0, zf_bf, 512, f1_a, nullptr, 0, EP_PRELU_ROWP_BF);
    gemm(zf_bf, 512, f1W2t, 512, 960, 512, 512, f1_b2, nullptr, 0,
         nullptr, 0, x_bf, 1024, nullptr, gv_p, 512, EP_SIGMUL_PACK_BF);
    gemm(gvp_bf, 512, f2W1t, 512, 960, 512, 512, f2_b1, nullptr, 0,
         nullptr, 0, zf_bf, 512, f2_a, nullptr, 0, EP_PRELU_ROWP_BF);
    gemm(zf_bf, 512, f2W2t, 512, 960, 512, 512, f2_b2, nullptr, 0,
         nullptr, 0, x_bf + 512, 1024, nullptr, gv_h, 512, EP_SIGMUL_PACK_BF);

    // ---- phase 3: BiLSTM ----
    for (int dir = 0; dir < 2; ++dir) {   // layer 0
        gemm(x_bf, 1024, Wih0b + (size_t)dir * 4096 * 1024, 1024, 960, 4096, 1024,
             bih0 + dir * 4096, nullptr, 0, xW, 4096, nullptr, 0, nullptr, nullptr, 0,
             EP_BIAS_F32);
        zf(cst, 64 * 1024); zb(h_bf, 64 * 1024);
        for (int s = 0; s < 15; ++s) {
            int t = dir ? (14 - s) : s;
            gemm(h_bf, 1024, Whh0b + (size_t)dir * 4096 * 1024, 1024, 64, 4096, 1024,
                 bhh0 + dir * 4096, xW + (size_t)t * 64 * 4096, 4096, gates, 4096,
                 nullptr, 0, nullptr, nullptr, 0, EP_BIAS_F32);
            k_lstm_gate<<<cdivu(64 * 1024, 256), 256, 0, stream>>>(
                gates, cst, h_bf, y1_bf + (size_t)t * 64 * 2048 + dir * 1024, 2048);
        }
    }
    for (int dir = 0; dir < 2; ++dir) {   // layer 1 (final h of each dir -> hcat)
        gemm(y1_bf, 2048, Wih1b + (size_t)dir * 4096 * 2048, 2048, 960, 4096, 2048,
             bih1 + dir * 4096, nullptr, 0, xW, 4096, nullptr, 0, nullptr, nullptr, 0,
             EP_BIAS_F32);
        zf(cst, 64 * 1024); zb(h_bf, 64 * 1024);
        for (int s = 0; s < 15; ++s) {
            int t = dir ? (14 - s) : s;
            gemm(h_bf, 1024, Whh1b + (size_t)dir * 4096 * 1024, 1024, 64, 4096, 1024,
                 bhh1 + dir * 4096, xW + (size_t)t * 64 * 4096, 4096, gates, 4096,
                 nullptr, 0, nullptr, nullptr, 0, EP_BIAS_F32);
            k_lstm_gate<<<cdivu(64 * 1024, 256), 256, 0, stream>>>(
                gates, cst, h_bf, hcat_bf + dir * 1024, 2048);
        }
    }

    // ---- phase 4: head ----
    gemm(hcat_bf, 2048, Wfct, 2048, 64, 512, 2048, bfc, nullptr, 0,
         nullptr, 0, f_bf, 512, a_fc, nullptr, 0, EP_PRELU_COL_BF);
    gemm(f_bf, 512, Wct, 512, 64, 120, 512, bc, nullptr, 0,
         (float*)d_out, 120, nullptr, 0, nullptr, nullptr, 0, EP_BIAS_F32);
}
